// EC_MoE_42752104464420
// MI455X (gfx1250) — compile-verified, matrix operand order checked by reading, not run on hardware
//
#include <hip/hip_runtime.h>
#include <hip/hip_bf16.h>
#include <stdint.h>

// ---------------------------------------------------------------------------
// Expert-choice MoE for MI455X (gfx1250), wave32 + WMMA bf16 + TDM async copy.
//   B=8192 tokens, D=1024, E=8 experts, K=2048 capacity, H=4*D=4096
// ---------------------------------------------------------------------------

typedef __attribute__((ext_vector_type(16))) __bf16 v16bf;
typedef __attribute__((ext_vector_type(8)))  __bf16 v8bf;
typedef __attribute__((ext_vector_type(8)))  float  v8f;
typedef unsigned int u32x4 __attribute__((ext_vector_type(4)));
typedef int          i32x8 __attribute__((ext_vector_type(8)));
typedef int          i32x4 __attribute__((ext_vector_type(4)));

#define BTOK 8192
#define DDIM 1024
#define NEXP 8
#define KCAP 2048
#define HDIM 4096
#define LDA  40   // padded LDS row stride (elements): 80B = 20 banks -> conflict-free b128 frag reads

// ---------------------------------------------------------------------------
// Kernel 1: router.  One wave (32 lanes) per token: logits = x_row @ Wr + br,
// softmax over E=8, write gate[B][E].
// ---------------------------------------------------------------------------
__global__ __launch_bounds__(256) void moe_router_kernel(
    const float* __restrict__ x, const float* __restrict__ Wr,
    const float* __restrict__ br, float* __restrict__ gate)
{
    const int wave = threadIdx.x >> 5;
    const int lane = threadIdx.x & 31;
    const int token = blockIdx.x * 8 + wave;
    if (token >= BTOK) return;

    float p[NEXP];
#pragma unroll
    for (int e = 0; e < NEXP; ++e) p[e] = 0.f;

    const float* xrow = x + (size_t)token * DDIM;
    for (int d = lane; d < DDIM; d += 32) {
        float xv = xrow[d];
        const float4* wr = (const float4*)(Wr + (size_t)d * NEXP);
        float4 w0 = wr[0], w1 = wr[1];
        p[0] += xv * w0.x; p[1] += xv * w0.y; p[2] += xv * w0.z; p[3] += xv * w0.w;
        p[4] += xv * w1.x; p[5] += xv * w1.y; p[6] += xv * w1.z; p[7] += xv * w1.w;
    }
#pragma unroll
    for (int e = 0; e < NEXP; ++e) {
#pragma unroll
        for (int off = 16; off > 0; off >>= 1)
            p[e] += __shfl_xor(p[e], off, 32);
    }
#pragma unroll
    for (int e = 0; e < NEXP; ++e) p[e] += br[e];

    float mx = p[0];
#pragma unroll
    for (int e = 1; e < NEXP; ++e) mx = fmaxf(mx, p[e]);
    float s = 0.f;
#pragma unroll
    for (int e = 0; e < NEXP; ++e) { p[e] = __expf(p[e] - mx); s += p[e]; }
    float inv = 1.f / s;

    if (lane == 0) {
        float4* g = (float4*)(gate + (size_t)token * NEXP);
        g[0] = make_float4(p[0] * inv, p[1] * inv, p[2] * inv, p[3] * inv);
        g[1] = make_float4(p[4] * inv, p[5] * inv, p[6] * inv, p[7] * inv);
    }
}

// ---------------------------------------------------------------------------
// Kernel 2: exact top-K per expert via 4-pass radix select on positive-float
// bits (monotonic), then compaction of (token index, gate weight) lists.
// One workgroup (256 threads) per expert.
// ---------------------------------------------------------------------------
__global__ __launch_bounds__(256) void moe_select_kernel(
    const float* __restrict__ gate, int* __restrict__ idxL, float* __restrict__ wgtL)
{
    const int e = blockIdx.x;
    const int tid = threadIdx.x;

    __shared__ unsigned sh_hist[256];
    __shared__ unsigned sh_prefix, sh_mask, sh_want, sh_cntG, sh_cntE;

    if (tid == 0) { sh_prefix = 0u; sh_mask = 0u; sh_want = KCAP; }
    __syncthreads();

#pragma unroll
    for (int shift = 24; shift >= 0; shift -= 8) {
        sh_hist[tid] = 0u;
        __syncthreads();
        unsigned prefix = sh_prefix, mask = sh_mask;
        for (int i = tid; i < BTOK; i += 256) {
            unsigned u = __float_as_uint(gate[(size_t)i * NEXP + e]);
            if ((u & mask) == prefix)
                atomicAdd(&sh_hist[(u >> shift) & 255u], 1u);
        }
        __syncthreads();
        if (tid == 0) {
            unsigned want = sh_want, cum = 0u;
            int b;
            for (b = 255; b >= 0; --b) {
                unsigned c = sh_hist[b];
                if (cum + c >= want) break;
                cum += c;
            }
            if (b < 0) b = 0;
            sh_prefix = prefix | ((unsigned)b << shift);
            sh_mask   = mask   | (255u << shift);
            sh_want   = want - cum;   // remaining slots inside chosen bin
        }
        __syncthreads();
    }

    if (tid == 0) { sh_cntG = 0u; sh_cntE = 0u; }
    __syncthreads();
    const unsigned T = sh_prefix;      // exact bits of K-th largest value
    const unsigned want = sh_want;     // number of ties to accept
    __syncthreads();

    for (int i = tid; i < BTOK; i += 256) {
        float g = gate[(size_t)i * NEXP + e];
        unsigned u = __float_as_uint(g);
        int slot = -1;
        if (u > T) {
            slot = (int)atomicAdd(&sh_cntG, 1u);                 // [0, K-want)
        } else if (u == T) {
            unsigned j = atomicAdd(&sh_cntE, 1u);
            if (j < want) slot = (int)(KCAP - want + j);         // [K-want, K)
        }
        if (slot >= 0) {
            idxL[(size_t)e * KCAP + slot] = i;
            wgtL[(size_t)e * KCAP + slot] = g;
        }
    }
}

// ---------------------------------------------------------------------------
// Fragment helpers: 16-bit A/B layout (ISA 7.12.2).  Lane L, hi = L>>4,
// m/n = L&15.  VGPRs 0..3 hold K = 8*hi + 0..7, VGPRs 4..7 hold K = 16+8*hi.
// With LDS rows stored K-contiguous, a fragment is two 16-byte chunks.
// ---------------------------------------------------------------------------
__device__ __forceinline__ v16bf load_frag(const __bf16* rowp, int hi)
{
    v8bf lo = *(const v8bf*)(rowp + 8 * hi);
    v8bf up = *(const v8bf*)(rowp + 16 + 8 * hi);
    v16bf f;
#pragma unroll
    for (int q = 0; q < 8; ++q) { f[q] = lo[q]; f[q + 8] = up[q]; }
    return f;
}

// ---------------------------------------------------------------------------
// Kernel 3: GEMM1 per expert:  H = relu(gather(x) @ W1[e] + b1[e])  -> bf16
//   tile 128x128, 8 waves (2 M x 4 N), each wave 64x32 = 4x2 WMMA tiles.
//   A is gathered fp32 (needs conversion) -> VALU path into LDS.
// ---------------------------------------------------------------------------
__global__ __launch_bounds__(256) void moe_gemm1_kernel(
    const float* __restrict__ x, const float* __restrict__ W1,
    const float* __restrict__ b1, const int* __restrict__ idxL,
    unsigned short* __restrict__ Hraw)
{
    const int e    = blockIdx.z;
    const int nBlk = blockIdx.x * 128;     // over HDIM
    const int mBlk = blockIdx.y * 128;     // over KCAP

    const float* W1e = W1 + (size_t)e * DDIM * HDIM;
    const float* b1e = b1 + (size_t)e * HDIM;
    const int*   idx = idxL + (size_t)e * KCAP + mBlk;
    __bf16*      Hb  = (__bf16*)Hraw + (size_t)e * KCAP * HDIM;

    __shared__ __bf16 As[128 * LDA];
    __shared__ __bf16 Bs[128 * LDA];

    const int tid  = threadIdx.x;
    const int lane = tid & 31;
    const int wave = tid >> 5;
    const int lm   = lane & 15;
    const int hi   = lane >> 4;
    const int mW   = (wave >> 2) * 64;
    const int nW   = (wave & 3) * 32;

    // loader mappings
    const int fA = tid & 7,  rA = tid >> 3;        // A: 8 float4 per row of 32
    const int n0 = (tid & 31) * 4, kp = tid >> 5;  // B: transpose

    v8f acc[4][2];
#pragma unroll
    for (int i = 0; i < 4; ++i)
#pragma unroll
        for (int j = 0; j < 2; ++j)
#pragma unroll
            for (int q = 0; q < 8; ++q) acc[i][j][q] = 0.f;

    for (int kB = 0; kB < DDIM; kB += 32) {
        // ---- A tile: gather 128 token rows, fp32 -> bf16, K-contiguous
#pragma unroll
        for (int m = 0; m < 4; ++m) {
            int r   = rA + m * 32;
            int row = idx[r];
            float4 v = *(const float4*)(x + (size_t)row * DDIM + kB + fA * 4);
            __bf16* dst = &As[r * LDA + fA * 4];
            dst[0] = (__bf16)v.x; dst[1] = (__bf16)v.y;
            dst[2] = (__bf16)v.z; dst[3] = (__bf16)v.w;
        }
        // ---- B tile: W1[kB..kB+32][nBlk..nBlk+128] transposed to [N][K]
#pragma unroll
        for (int kk = 0; kk < 4; ++kk) {
            int k = kp + kk * 8;
            float4 v = *(const float4*)(W1e + (size_t)(kB + k) * HDIM + nBlk + n0);
            Bs[(n0 + 0) * LDA + k] = (__bf16)v.x;
            Bs[(n0 + 1) * LDA + k] = (__bf16)v.y;
            Bs[(n0 + 2) * LDA + k] = (__bf16)v.z;
            Bs[(n0 + 3) * LDA + k] = (__bf16)v.w;
        }
        if (kB + 32 < DDIM)
            __builtin_prefetch(W1e + (size_t)(kB + 32 + kp) * HDIM + nBlk + n0, 0, 1);
        __syncthreads();

        v16bf a[4], b[2];
#pragma unroll
        for (int i = 0; i < 4; ++i)
            a[i] = load_frag(&As[(mW + i * 16 + lm) * LDA], hi);
#pragma unroll
        for (int j = 0; j < 2; ++j)
            b[j] = load_frag(&Bs[(nW + j * 16 + lm) * LDA], hi);

#pragma unroll
        for (int i = 0; i < 4; ++i)
#pragma unroll
            for (int j = 0; j < 2; ++j)
                acc[i][j] = __builtin_amdgcn_wmma_f32_16x16x32_bf16(
                    false, a[i], false, b[j], (short)0, acc[i][j], false, false);
        __syncthreads();
    }

    // ---- epilogue: bias + relu -> bf16 H
#pragma unroll
    for (int j = 0; j < 2; ++j) {
        int nCol = nBlk + nW + j * 16 + lm;
        float bias = b1e[nCol];
#pragma unroll
        for (int i = 0; i < 4; ++i) {
            int mBase = mW + i * 16 + 8 * hi;
#pragma unroll
            for (int r = 0; r < 8; ++r) {
                float h = acc[i][j][r] + bias;
                h = h > 0.f ? h : 0.f;
                Hb[(size_t)(mBlk + mBase + r) * HDIM + nCol] = (__bf16)h;
            }
        }
    }
}

// ---------------------------------------------------------------------------
// Kernel 4: GEMM2 per expert:  y = H @ W2[e] + b2[e];  out[token] += w * y
//   A tile (H, already bf16) is fetched by the Tensor Data Mover:
//   one wave issues TENSOR_LOAD_TO_LDS with a 2D D# (tile 32x128, 2B elems),
//   using TDM padding (16 DWORDs data, 4 DWORDs pad) to produce the same
//   80-byte LDS row stride (LDA=40) the WMMA fragment loads expect.
// ---------------------------------------------------------------------------
__global__ __launch_bounds__(256) void moe_gemm2_kernel(
    const unsigned short* __restrict__ Hraw, const float* __restrict__ W2,
    const float* __restrict__ b2, const int* __restrict__ idxL,
    const float* __restrict__ wgtL, float* __restrict__ out)
{
    const int e    = blockIdx.z;
    const int nBlk = blockIdx.x * 128;     // over DDIM
    const int mBlk = blockIdx.y * 128;     // over KCAP

    const __bf16* Hb  = (const __bf16*)Hraw + (size_t)e * KCAP * HDIM;
    const float*  W2e = W2 + (size_t)e * HDIM * DDIM;
    const float*  b2e = b2 + (size_t)e * DDIM;
    const int*    idx = idxL + (size_t)e * KCAP + mBlk;
    const float*  wgt = wgtL + (size_t)e * KCAP + mBlk;

    __shared__ __bf16 As[128 * LDA];
    __shared__ __bf16 Bs[128 * LDA];

    const int tid  = threadIdx.x;
    const int lane = tid & 31;
    const int wave = tid >> 5;
    const int lm   = lane & 15;
    const int hi   = lane >> 4;
    const int mW   = (wave >> 2) * 64;
    const int nW   = (wave & 3) * 32;

    const int n0 = (tid & 31) * 4, kp = tid >> 5;  // B: transpose

    // --- TDM descriptor pieces that are loop-invariant (uniform, -> SGPRs)
    const unsigned ldsA = (unsigned)(size_t)(&As[0]);
    const unsigned long long gbase =
        (unsigned long long)(size_t)Hb + (unsigned long long)mBlk * HDIM * 2ull;

    i32x8 g1;
    g1[0] = (1 << 16)      // data_size = 1 -> 2 bytes
          | (1 << 20)      // pad_enable
          | (3 << 22)      // pad_interval: pad after 16 DWORDs (one 64B row)
          | (3 << 25);     // pad_amount: 4 DWORDs (16B)  => row stride 80B
    g1[1] = (int)((unsigned)HDIM << 16);   // tensor_dim0 = 4096 (low 16 in [31:16])
    g1[2] = (int)((unsigned)KCAP << 16);   // tensor_dim1 = 2048
    g1[3] = (int)(32u << 16);              // tile_dim0 = 32
    g1[4] = 128;                           // tile_dim1 = 128
    g1[5] = HDIM;                          // tensor_dim0_stride = 4096
    g1[6] = 0;
    g1[7] = 0;
    const i32x4 gz4 = {0, 0, 0, 0};
    const i32x8 gz8 = {0, 0, 0, 0, 0, 0, 0, 0};

    v8f acc[4][2];
#pragma unroll
    for (int i = 0; i < 4; ++i)
#pragma unroll
        for (int j = 0; j < 2; ++j)
#pragma unroll
            for (int q = 0; q < 8; ++q) acc[i][j][q] = 0.f;

    for (int kB = 0; kB < HDIM; kB += 32) {
        // ---- A tile via Tensor Data Mover (async DMA, one wave issues)
        if (wave == 0) {
            unsigned long long gaddr = gbase + (unsigned long long)kB * 2ull;
            u32x4 g0;
            g0[0] = 1u;                                   // count = 1, user mode
            g0[1] = ldsA;                                 // lds_addr
            g0[2] = (unsigned)gaddr;                      // global_addr[31:0]
            g0[3] = (unsigned)((gaddr >> 32) & 0x01FFFFFFull)
                  | (2u << 30);                           // type = 2 ("image")
            __builtin_amdgcn_tensor_load_to_lds(g0, g1, gz4, gz4, gz8, 0);
        }
        // ---- B tile: W2[kB..kB+32][nBlk..nBlk+128] transposed to [N][K]
        //      (fp32 -> bf16 conversion, overlapped with the TDM transfer)
#pragma unroll
        for (int kk = 0; kk < 4; ++kk) {
            int k = kp + kk * 8;
            float4 v = *(const float4*)(W2e + (size_t)(kB + k) * DDIM + nBlk + n0);
            Bs[(n0 + 0) * LDA + k] = (__bf16)v.x;
            Bs[(n0 + 1) * LDA + k] = (__bf16)v.y;
            Bs[(n0 + 2) * LDA + k] = (__bf16)v.z;
            Bs[(n0 + 3) * LDA + k] = (__bf16)v.w;
        }
        if (kB + 32 < HDIM)
            __builtin_prefetch(W2e + (size_t)(kB + 32 + kp) * DDIM + nBlk + n0, 0, 1);
        if (wave == 0)
            __builtin_amdgcn_s_wait_tensorcnt(0);
        __syncthreads();

        v16bf a[4], b[2];
#pragma unroll
        for (int i = 0; i < 4; ++i)
            a[i] = load_frag(&As[(mW + i * 16 + lm) * LDA], hi);
#pragma unroll
        for (int j = 0; j < 2; ++j)
            b[j] = load_frag(&Bs[(nW + j * 16 + lm) * LDA], hi);

#pragma unroll
        for (int i = 0; i < 4; ++i)
#pragma unroll
            for (int j = 0; j < 2; ++j)
                acc[i][j] = __builtin_amdgcn_wmma_f32_16x16x32_bf16(
                    false, a[i], false, b[j], (short)0, acc[i][j], false, false);
        __syncthreads();
    }

    // ---- epilogue: bias, weight, scatter-add into out
    float bias0 = b2e[nBlk + nW + lm];
    float bias1 = b2e[nBlk + nW + 16 + lm];
#pragma unroll
    for (int i = 0; i < 4; ++i) {
        int mBase = mW + i * 16 + 8 * hi;
#pragma unroll
        for (int r = 0; r < 8; ++r) {
            int mRow  = mBase + r;
            int token = idx[mRow];
            float wm  = wgt[mRow];
            float* orow = out + (size_t)token * DDIM + nBlk;
            float y0 = acc[i][0][r] + bias0;
            float y1 = acc[i][1][r] + bias1;
            unsafeAtomicAdd(&orow[nW + lm],      wm * y0);
            unsafeAtomicAdd(&orow[nW + 16 + lm], wm * y1);
        }
    }
}

// ---------------------------------------------------------------------------
// Host launcher
// ---------------------------------------------------------------------------
extern "C" void kernel_launch(void* const* d_in, const int* in_sizes, int n_in,
                              void* d_out, int out_size, void* d_ws, size_t ws_size,
                              hipStream_t stream)
{
    (void)in_sizes; (void)n_in; (void)out_size; (void)ws_size;

    const float* x  = (const float*)d_in[0];   // [B, D]
    const float* Wr = (const float*)d_in[1];   // [D, E]
    const float* br = (const float*)d_in[2];   // [E]
    const float* W1 = (const float*)d_in[3];   // [E, D, 4D]
    const float* b1 = (const float*)d_in[4];   // [E, 4D]
    const float* W2 = (const float*)d_in[5];   // [E, 4D, D]
    const float* b2 = (const float*)d_in[6];   // [E, D]
    float* out = (float*)d_out;                // [B, D]

    char* ws = (char*)d_ws;
    float*          gate = (float*)ws;                                   // 256 KB
    int*            idxL = (int*)  (ws + (size_t)BTOK * NEXP * 4);       // 64 KB
    float*          wgtL = (float*)(ws + (size_t)BTOK * NEXP * 4 + (size_t)NEXP * KCAP * 4);
    unsigned short* Hbuf = (unsigned short*)(ws + (size_t)BTOK * NEXP * 4
                                               + (size_t)NEXP * KCAP * 8);  // 128 MB bf16

    (void)hipMemsetAsync(d_out, 0, (size_t)BTOK * DDIM * sizeof(float), stream);

    moe_router_kernel<<<BTOK / 8, 256, 0, stream>>>(x, Wr, br, gate);
    moe_select_kernel<<<NEXP, 256, 0, stream>>>(gate, idxL, wgtL);
    moe_gemm1_kernel<<<dim3(HDIM / 128, KCAP / 128, NEXP), 256, 0, stream>>>(
        x, W1, b1, idxL, Hbuf);
    moe_gemm2_kernel<<<dim3(DDIM / 128, KCAP / 128, NEXP), 256, 0, stream>>>(
        Hbuf, W2, b2, idxL, wgtL, out);
}
